// HeterogeneousNormalizedAttentionBlock_19628000543021
// MI455X (gfx1250) — matrix operand order repaired and checked by reading, not compiled, not run on hardware
//
#include <hip/hip_runtime.h>
#include <hip/hip_bf16.h>
#include <stdint.h>
#include <math.h>

// ---------------------------------------------------------------------------
// MI455X (gfx1250) implementation.
// 34 GEMMs of (32768 x 256) @ (256 x 256)^T dominate (~146 GFLOP); all run on
// v_wmma_f32_16x16x32_bf16 (wave32). Intermediate activations are kept in
// bf16 to halve HBM traffic (23.3 TB/s roofline); the residual accumulator
// (d_out) stays fp32. bf16 A-tiles are staged into LDS with
// global_load_async_to_lds_b128 (ASYNCcnt-tracked DMA, no VGPR round trip);
// fp32 weights are converted in-register while the async DMA is in flight.
// ---------------------------------------------------------------------------

#define B_    4
#define LQ_   8192
#define LK_   8192
#define E_    256
#define H_    8
#define D_    32
#define NF_   2
#define NE_   4
#define HID_  256
#define NTOK  (B_*LQ_)   // 32768 tokens (also B_*LK_)

typedef __bf16 bf16_t;
typedef bf16_t v16bf __attribute__((ext_vector_type(16)));
typedef bf16_t v8bf  __attribute__((ext_vector_type(8)));
typedef bf16_t v4bf  __attribute__((ext_vector_type(4)));
typedef float  v8f   __attribute__((ext_vector_type(8)));

enum { MODE_BIAS = 0, MODE_GELU = 1, MODE_SCORE = 2 };

// Generic LDS pointer -> 32-bit LDS byte address (ISA 10.2: low 32 bits of the address)
__device__ __forceinline__ unsigned lds_addr32(const void* p)
{
    return (unsigned)(uintptr_t)p;
}

// ---------------------------------------------------------------------------
// GEMM: Out(M,256) = epilogue( X(M,256) @ W(256,256)^T + bias )
// grid = (M/128, 2), block = 256 threads (8 waves in a 4x2 wave grid).
// Each wave owns a 32x64 tile = 2x4 accumulators of 16x16 f32.
// K=256 walked in 4 chunks of 64 staged through LDS as bf16.
// TIN = bf16: A staged via async global->LDS DMA. TIN = float: convert path.
// ---------------------------------------------------------------------------
template <int MODE, typename TIN, typename TOUT>
__global__ __launch_bounds__(256)
void gemm256_kernel(const TIN* __restrict__ X, const float* __restrict__ W,
                    const float* __restrict__ bias, TOUT* __restrict__ Out,
                    const float* __restrict__ scores, int expert)
{
    __shared__ __align__(16) bf16_t As[128][72];   // 128 rows x 64 k (+pad)
    __shared__ __align__(16) bf16_t Bs[128][72];

    const int tid  = threadIdx.x;
    const int lane = tid & 31;
    const int wave = tid >> 5;
    const int wm   = wave & 3;         // wave row group (32 rows each)
    const int wn   = wave >> 2;        // wave col group (64 cols each)
    const int rowBase = blockIdx.x * 128;
    const int colBase = blockIdx.y * 128;
    const int half = lane >> 4;        // lane half (WMMA layout)
    const int lr   = lane & 15;
    const uint64_t xbase = (uint64_t)X;

    v8f acc[2][4];
#pragma unroll
    for (int i = 0; i < 2; ++i)
#pragma unroll
        for (int j = 0; j < 4; ++j) acc[i][j] = (v8f)0.0f;

    for (int kc = 0; kc < 4; ++kc) {
        const int kBase = kc * 64;
        __syncthreads();

        if constexpr (sizeof(TIN) == 2) {
            // A tile is already bf16: async DMA straight into LDS.
            // 128 rows x 128 bytes = 1024 x 16B segments; 4 per thread.
#pragma unroll
            for (int i = 0; i < 4; ++i) {
                const int linear = tid + i * 256;
                const int r = linear >> 3;
                const int c = linear & 7;
                const unsigned lofs = lds_addr32(&As[r][c * 8]);
                const unsigned goff =
                    (unsigned)((((rowBase + r) * 256 + kBase + c * 8)) * 2);
                asm volatile("global_load_async_to_lds_b128 %0, %1, %2"
                             :: "v"(lofs), "v"(goff), "s"(xbase) : "memory");
            }
        } else {
            // fp32 A: load + convert to bf16 in registers.
#pragma unroll
            for (int i = 0; i < 8; ++i) {
                const int linear = tid + i * 256;   // float4 units, 2048 total
                const int r  = linear >> 4;
                const int c4 = linear & 15;
                const float4 fa = *(const float4*)((const float*)X +
                                   (size_t)(rowBase + r) * 256 + kBase + c4 * 4);
                union { bf16_t h[4]; uint64_t u; } pa;
                pa.h[0] = (bf16_t)fa.x; pa.h[1] = (bf16_t)fa.y;
                pa.h[2] = (bf16_t)fa.z; pa.h[3] = (bf16_t)fa.w;
                *(uint64_t*)&As[r][c4 * 4] = pa.u;
            }
        }

        // B (weights, fp32, L2-resident): load + convert while DMA in flight.
#pragma unroll
        for (int i = 0; i < 8; ++i) {
            const int linear = tid + i * 256;
            const int r  = linear >> 4;
            const int c4 = linear & 15;
            const float4 fb = *(const float4*)(W + (size_t)(colBase + r) * 256 +
                                               kBase + c4 * 4);
            union { bf16_t h[4]; uint64_t u; } pb;
            pb.h[0] = (bf16_t)fb.x; pb.h[1] = (bf16_t)fb.y;
            pb.h[2] = (bf16_t)fb.z; pb.h[3] = (bf16_t)fb.w;
            *(uint64_t*)&Bs[r][c4 * 4] = pb.u;
        }

        if constexpr (sizeof(TIN) == 2) {
            asm volatile("s_wait_asynccnt 0x0" ::: "memory");
        }
        __syncthreads();

#pragma unroll
        for (int s = 0; s < 2; ++s) {             // two K=32 WMMA steps / chunk
            const int k0 = s * 32;
            v16bf afr[2], bfr[4];
#pragma unroll
            for (int mt = 0; mt < 2; ++mt) {
                // A layout (16-bit, 16x32): lane half h: elems 0-7 -> k0+8h+i,
                // elems 8-15 -> k0+16+8h+(i-8)
                const bf16_t* p = &As[wm * 32 + mt * 16 + lr][k0 + half * 8];
                v8bf lo = *(const v8bf*)p;
                v8bf hi = *(const v8bf*)(p + 16);
                afr[mt] = __builtin_shufflevector(lo, hi, 0, 1, 2, 3, 4, 5, 6, 7,
                                                  8, 9, 10, 11, 12, 13, 14, 15);
            }
#pragma unroll
            for (int nt = 0; nt < 4; ++nt) {
                // B layout (16-bit, 32x16): lane half h: elems 0-7 -> k0+8h+i,
                // elems 8-15 -> k0+16+8h+(i-8)
                const bf16_t* p = &Bs[wn * 64 + nt * 16 + lr][k0 + half * 8];
                v8bf lo = *(const v8bf*)p;
                v8bf hi = *(const v8bf*)(p + 16);
                bfr[nt] = __builtin_shufflevector(lo, hi, 0, 1, 2, 3, 4, 5, 6, 7,
                                                  8, 9, 10, 11, 12, 13, 14, 15);
            }
#pragma unroll
            for (int mt = 0; mt < 2; ++mt)
#pragma unroll
                for (int nt = 0; nt < 4; ++nt)
                    acc[mt][nt] = __builtin_amdgcn_wmma_f32_16x16x32_bf16(
                        false, afr[mt], false, bfr[nt], (short)0, acc[mt][nt],
                        false, false);
        }
    }

    // Epilogue. C/D layout: elem r -> M = r + 8*half, lane%16 -> N.
#pragma unroll
    for (int mt = 0; mt < 2; ++mt) {
#pragma unroll
        for (int nt = 0; nt < 4; ++nt) {
            const int n = colBase + wn * 64 + nt * 16 + lr;
            const float bv = bias[n];
#pragma unroll
            for (int r = 0; r < 8; ++r) {
                const int m = rowBase + wm * 32 + mt * 16 + half * 8 + r;
                float v = acc[mt][nt][r] + bv;
                if constexpr (MODE == MODE_GELU) {
                    v = 0.5f * v * (1.0f + erff(v * 0.70710678118654752f));
                }
                if constexpr (MODE == MODE_SCORE) {
                    const float sc = scores[(size_t)m * NE_ + expert];
                    Out[(size_t)m * 256 + n] += sc * v;   // fp32 accumulate
                } else {
                    Out[(size_t)m * 256 + n] = (TOUT)v;
                }
            }
        }
    }
}

// ---------------------------------------------------------------------------
// Per-head softmax over D=32 (one wave32 per (token, head)) — in place, bf16.
// ---------------------------------------------------------------------------
__global__ __launch_bounds__(256)
void softmax_head_kernel(bf16_t* __restrict__ buf)
{
    const int wid  = blockIdx.x * 8 + (threadIdx.x >> 5);
    const int lane = threadIdx.x & 31;
    const int m = wid >> 3;   // / H_
    const int h = wid & 7;
    const size_t idx = (size_t)m * E_ + h * D_ + lane;
    const float x = (float)buf[idx];
    float mx = x;
#pragma unroll
    for (int off = 16; off > 0; off >>= 1) mx = fmaxf(mx, __shfl_xor(mx, off, 32));
    const float e = expf(x - mx);
    float s = e;
#pragma unroll
    for (int off = 16; off > 0; off >>= 1) s += __shfl_xor(s, off, 32);
    buf[idx] = (bf16_t)(e / s);
}

// ---------------------------------------------------------------------------
// tmp[b,h,d1,d2] += sum_l k[l,d1] * v[l,d2];  ksum[b,h,d1] += sum_l k[l,d1]
// grid = (B*H, 16 l-chunks of 512), block = 256. fp32 atomics accumulate.
// ---------------------------------------------------------------------------
__global__ __launch_bounds__(256)
void kv_outer_kernel(const bf16_t* __restrict__ Kb, const bf16_t* __restrict__ Vb,
                     float* __restrict__ tmp, float* __restrict__ ksum)
{
    __shared__ float ks[32][36];
    __shared__ float vs[32][36];
    const int tid   = threadIdx.x;
    const int combo = blockIdx.x;           // b*H + h
    const int b = combo >> 3;
    const int h = combo & 7;
    const int lBase = blockIdx.y * 512;
    const int d1  = tid >> 3;               // 0..31 (also the LDS load row)
    const int d2b = (tid & 7) * 4;
    float a0 = 0.f, a1 = 0.f, a2 = 0.f, a3 = 0.f, ksacc = 0.f;

    for (int c = 0; c < 16; ++c) {
        __syncthreads();
        const size_t row = (size_t)(b * LK_ + lBase + c * 32 + d1) * E_ + h * D_;
        const v4bf kk = *(const v4bf*)(Kb + row + d2b);
        const v4bf vv = *(const v4bf*)(Vb + row + d2b);
        ks[d1][d2b + 0] = (float)kk[0]; ks[d1][d2b + 1] = (float)kk[1];
        ks[d1][d2b + 2] = (float)kk[2]; ks[d1][d2b + 3] = (float)kk[3];
        vs[d1][d2b + 0] = (float)vv[0]; vs[d1][d2b + 1] = (float)vv[1];
        vs[d1][d2b + 2] = (float)vv[2]; vs[d1][d2b + 3] = (float)vv[3];
        __syncthreads();
#pragma unroll 8
        for (int l = 0; l < 32; ++l) {
            const float kv = ks[l][d1];
            a0 += kv * vs[l][d2b + 0];
            a1 += kv * vs[l][d2b + 1];
            a2 += kv * vs[l][d2b + 2];
            a3 += kv * vs[l][d2b + 3];
        }
        if (tid < 32) {
#pragma unroll 8
            for (int l = 0; l < 32; ++l) ksacc += ks[l][tid];
        }
    }
    float* t = tmp + (size_t)combo * (D_ * D_) + d1 * D_ + d2b;
    atomicAdd(t + 0, a0); atomicAdd(t + 1, a1);
    atomicAdd(t + 2, a2); atomicAdd(t + 3, a3);
    if (tid < 32) atomicAdd(ksum + combo * D_ + tid, ksacc);
}

// ---------------------------------------------------------------------------
// Outp = q + invF * sum_f alpha_f * (q @ tmp_f);  alpha_f = 1/(q . ksum_f)
// One wave32 per (token, head); lane = output dim dv.
// ---------------------------------------------------------------------------
__global__ __launch_bounds__(256)
void attn_apply_kernel(const bf16_t* __restrict__ Q, const float* __restrict__ tmp,
                       const float* __restrict__ ksum, bf16_t* __restrict__ Outp,
                       int F, float invF)
{
    const int wid  = blockIdx.x * 8 + (threadIdx.x >> 5);
    const int lane = threadIdx.x & 31;
    const int m = wid >> 3;
    const int h = wid & 7;
    const int b = m / LQ_;
    const float qv = (float)Q[(size_t)m * E_ + h * D_ + lane];
    float outAcc = 0.f;
    for (int f = 0; f < F; ++f) {
        const int base = (f * B_ + b) * H_ + h;
        float s = qv * ksum[base * D_ + lane];
#pragma unroll
        for (int off = 16; off > 0; off >>= 1) s += __shfl_xor(s, off, 32);
        const float alpha = 1.0f / s;
        const float* tp = tmp + (size_t)base * (D_ * D_) + lane;
        float a = 0.f;
#pragma unroll
        for (int d1 = 0; d1 < 32; ++d1) a += __shfl(qv, d1, 32) * tp[(size_t)d1 * D_];
        outAcc += alpha * a;
    }
    Outp[(size_t)m * E_ + h * D_ + lane] = (bf16_t)(qv + invF * outAcc);
}

__global__ __launch_bounds__(256)
void zero_kernel(float* __restrict__ p, int n)
{
    const int i = blockIdx.x * 256 + threadIdx.x;
    if (i < n) p[i] = 0.f;
}

// ---------------------------------------------------------------------------
extern "C" void kernel_launch(void* const* d_in, const int* in_sizes, int n_in,
                              void* d_out, int out_size, void* d_ws, size_t ws_size,
                              hipStream_t stream)
{
    const float* scores = (const float*)d_in[0];
    const float* query  = (const float*)d_in[1];
    const float* infun  = (const float*)d_in[2];
    const float* ca_Wq = (const float*)d_in[3];  const float* ca_bq = (const float*)d_in[4];
    const float* ca_Wk = (const float*)d_in[5];  const float* ca_bk = (const float*)d_in[6];
    const float* ca_Wv = (const float*)d_in[7];  const float* ca_bv = (const float*)d_in[8];
    const float* ca_Wo = (const float*)d_in[9];  const float* ca_bo = (const float*)d_in[10];
    const float* sa_Wq = (const float*)d_in[11]; const float* sa_bq = (const float*)d_in[12];
    const float* sa_Wk = (const float*)d_in[13]; const float* sa_bk = (const float*)d_in[14];
    const float* sa_Wv = (const float*)d_in[15]; const float* sa_bv = (const float*)d_in[16];
    const float* sa_Wo = (const float*)d_in[17]; const float* sa_bo = (const float*)d_in[18];
    const float* f1_W1 = (const float*)d_in[19]; const float* f1_b1 = (const float*)d_in[20];
    const float* f1_W2 = (const float*)d_in[21]; const float* f1_b2 = (const float*)d_in[22];
    const float* f1_W3 = (const float*)d_in[23]; const float* f1_b3 = (const float*)d_in[24];
    const float* f2_W1 = (const float*)d_in[25]; const float* f2_b1 = (const float*)d_in[26];
    const float* f2_W2 = (const float*)d_in[27]; const float* f2_b2 = (const float*)d_in[28];
    const float* f2_W3 = (const float*)d_in[29]; const float* f2_b3 = (const float*)d_in[30];

    float* out = (float*)d_out;
    const size_t SZ = (size_t)NTOK * E_;    // elements per activation buffer
    bf16_t* wsb = (bf16_t*)d_ws;
    bf16_t* bufQ = wsb;                      // q / h1 scratch (bf16, 16 MB)
    bf16_t* bufK = wsb + SZ;                 // k / merged attn out
    bf16_t* bufV = wsb + 2 * SZ;             // v / h2 scratch
    bf16_t* bufX = wsb + 3 * SZ;             // ca / sa activation
    bf16_t* bufH1 = bufQ;
    bf16_t* bufH2 = bufV;
    float* tmpB = (float*)(wsb + 4 * SZ);    // (NF,B,H,32,32) fp32
    float* ksum = tmpB + (size_t)NF_ * B_ * H_ * D_ * D_;  // (NF,B,H,32)

    const dim3 blk(256);
    const dim3 gGemm(NTOK / 128, 2);
    const dim3 gSm(NTOK * H_ / 8);

    // residual base: out = query (fp32)
    hipMemcpyAsync(out, query, SZ * sizeof(float), hipMemcpyDeviceToDevice, stream);

    // ======================= cross attention ===============================
    gemm256_kernel<MODE_BIAS, float, bf16_t><<<gGemm, blk, 0, stream>>>(
        query, ca_Wq, ca_bq, bufQ, nullptr, 0);
    softmax_head_kernel<<<gSm, blk, 0, stream>>>(bufQ);

    const int tmpTotal = NF_ * B_ * H_ * D_ * D_ + NF_ * B_ * H_ * D_;
    zero_kernel<<<dim3((tmpTotal + 255) / 256), blk, 0, stream>>>(tmpB, tmpTotal);

    for (int f = 0; f < NF_; ++f) {
        const float* Xf = infun + (size_t)f * NTOK * E_;
        gemm256_kernel<MODE_BIAS, float, bf16_t><<<gGemm, blk, 0, stream>>>(
            Xf, ca_Wk + (size_t)f * E_ * E_, ca_bk + f * E_, bufK, nullptr, 0);
        softmax_head_kernel<<<gSm, blk, 0, stream>>>(bufK);
        gemm256_kernel<MODE_BIAS, float, bf16_t><<<gGemm, blk, 0, stream>>>(
            Xf, ca_Wv + (size_t)f * E_ * E_, ca_bv + f * E_, bufV, nullptr, 0);
        kv_outer_kernel<<<dim3(B_ * H_, 16), blk, 0, stream>>>(
            bufK, bufV, tmpB + (size_t)f * B_ * H_ * D_ * D_, ksum + f * B_ * H_ * D_);
    }
    attn_apply_kernel<<<gSm, blk, 0, stream>>>(bufQ, tmpB, ksum, bufK, NF_, 1.0f / NF_);
    gemm256_kernel<MODE_BIAS, bf16_t, bf16_t><<<gGemm, blk, 0, stream>>>(
        bufK, ca_Wo, ca_bo, bufX, nullptr, 0);

    // expert FFN 1, score-weighted accumulate into out (fp32)
    for (int e = 0; e < NE_; ++e) {
        gemm256_kernel<MODE_GELU, bf16_t, bf16_t><<<gGemm, blk, 0, stream>>>(
            bufX, f1_W1 + (size_t)e * HID_ * E_, f1_b1 + e * HID_, bufH1, nullptr, 0);
        gemm256_kernel<MODE_GELU, bf16_t, bf16_t><<<gGemm, blk, 0, stream>>>(
            bufH1, f1_W2 + (size_t)e * HID_ * HID_, f1_b2 + e * HID_, bufH2, nullptr, 0);
        gemm256_kernel<MODE_SCORE, bf16_t, float><<<gGemm, blk, 0, stream>>>(
            bufH2, f1_W3 + (size_t)e * HID_ * HID_, f1_b3 + e * HID_, out, scores, e);
    }

    // ======================= self attention ================================
    gemm256_kernel<MODE_BIAS, float, bf16_t><<<gGemm, blk, 0, stream>>>(
        out, sa_Wq, sa_bq, bufQ, nullptr, 0);
    softmax_head_kernel<<<gSm, blk, 0, stream>>>(bufQ);
    gemm256_kernel<MODE_BIAS, float, bf16_t><<<gGemm, blk, 0, stream>>>(
        out, sa_Wk, sa_bk, bufK, nullptr, 0);
    softmax_head_kernel<<<gSm, blk, 0, stream>>>(bufK);
    gemm256_kernel<MODE_BIAS, float, bf16_t><<<gGemm, blk, 0, stream>>>(
        out, sa_Wv, sa_bv, bufV, nullptr, 0);

    zero_kernel<<<dim3((B_ * H_ * D_ * D_ + 255) / 256), blk, 0, stream>>>(
        tmpB, B_ * H_ * D_ * D_);
    zero_kernel<<<dim3((B_ * H_ * D_ + 255) / 256), blk, 0, stream>>>(
        ksum, B_ * H_ * D_);
    kv_outer_kernel<<<dim3(B_ * H_, 16), blk, 0, stream>>>(bufK, bufV, tmpB, ksum);
    attn_apply_kernel<<<gSm, blk, 0, stream>>>(bufQ, tmpB, ksum, bufK, 1, 1.0f);
    gemm256_kernel<MODE_BIAS, bf16_t, bf16_t><<<gGemm, blk, 0, stream>>>(
        bufK, sa_Wo, sa_bo, bufX, nullptr, 0);

    // expert FFN 2, score-weighted accumulate into out (fp32)
    for (int e = 0; e < NE_; ++e) {
        gemm256_kernel<MODE_GELU, bf16_t, bf16_t><<<gGemm, blk, 0, stream>>>(
            bufX, f2_W1 + (size_t)e * HID_ * E_, f2_b1 + e * HID_, bufH1, nullptr, 0);
        gemm256_kernel<MODE_GELU, bf16_t, bf16_t><<<gGemm, blk, 0, stream>>>(
            bufH1, f2_W2 + (size_t)e * HID_ * HID_, f2_b2 + e * HID_, bufH2, nullptr, 0);
        gemm256_kernel<MODE_SCORE, bf16_t, float><<<gGemm, blk, 0, stream>>>(
            bufH2, f2_W3 + (size_t)e * HID_ * HID_, f2_b3 + e * HID_, out, scores, e);
    }
}
